// SuperGlue_13932873909019
// MI455X (gfx1250) — matrix operand order repaired and checked by reading, not compiled, not run on hardware
//
#include <hip/hip_runtime.h>
#include <hip/hip_bf16.h>

// ---------------------------------------------------------------------------
// Types / WMMA helpers (gfx1250, wave32)
// ---------------------------------------------------------------------------
typedef __attribute__((ext_vector_type(16))) _Float16 v16h;
typedef __attribute__((ext_vector_type(8)))  _Float16 v8h;
typedef __attribute__((ext_vector_type(8)))  float    v8f;

union V16H { v16h v; v8h h8[2]; };

// A fragment: 16(M) x 32(K) f16. Lanes 0-15 -> M=lane, halves = K[0..7],K[16..23];
// lanes 16-31 -> M=lane-16, halves = K[8..15],K[24..31]. LDS layout [m][k], ldh % 8 == 0.
__device__ __forceinline__ v16h ld_frag_a(const _Float16* base, int ldh, int mbase,
                                          int kbase, int lane) {
  int lr = lane & 15, hi = lane >> 4;
  const _Float16* p = base + (size_t)(mbase + lr) * ldh + kbase + hi * 8;
  V16H r;
  r.h8[0] = *(const v8h*)(p);
  r.h8[1] = *(const v8h*)(p + 16);
  return r.v;
}

// B fragment: 32(K) x 16(N) f16. Lanes 0-15: N=lane, halves = K[0..7],K[16..23];
// lanes 16-31: N=lane-16, halves = K[8..15],K[24..31]. LDS layout [n][k] (K-contiguous).
__device__ __forceinline__ v16h ld_frag_b(const _Float16* base, int ldh, int nbase,
                                          int kbase, int lane) {
  int lr = lane & 15, hi = lane >> 4;
  const _Float16* p = base + (size_t)(nbase + lr) * ldh + kbase + hi * 8;
  V16H r;
  r.h8[0] = *(const v8h*)(p);
  r.h8[1] = *(const v8h*)(p + 16);
  return r.v;
}

#define WMMA_F16(a, b, c) \
  __builtin_amdgcn_wmma_f32_16x16x32_f16(false, (a), false, (b), (short)0, (c), false, false)

// Async DMA global->LDS (gfx1250; ASYNCcnt-tracked). LDS byte address is the
// low 32 bits of the flat shared-memory pointer (ISA: LDS aperture addr[31:0]).
__device__ __forceinline__ void async_g2l_b128(void* lds, const void* g) {
  unsigned l = (unsigned)(unsigned long long)lds;
  unsigned long long ga = (unsigned long long)g;
  asm volatile("global_load_async_to_lds_b128 %0, %1, off" :: "v"(l), "v"(ga) : "memory");
}
__device__ __forceinline__ void wait_async0() {
  asm volatile("s_wait_asynccnt 0x0" ::: "memory");
}

#define G_LD 48   // LDS row stride (halves) for 32-deep K tiles
#define AT_LD 80  // LDS row stride (halves) for 64-wide attention tiles

// ---------------------------------------------------------------------------
// fp32 -> fp16 weight pre-conversion (amortized once per call; weights are
// L2-resident and reused by every tile -> halves weight traffic, removes
// per-tile cvt VALU from the GEMM hot loop, enables async DMA staging)
// ---------------------------------------------------------------------------
__global__ void cvt16_k(const float* __restrict__ src, _Float16* __restrict__ dst, int n)
{
  int i = blockIdx.x * blockDim.x + threadIdx.x;
  if (i < n) dst[i] = (_Float16)src[i];
}

// ---------------------------------------------------------------------------
// Fused GEMM: Y[b,o,n] = act( sum_i W[o,i]*X[b,i,n] + bias[o] ) (+ residual)
// W in fp16; A-tiles staged by async DMA; double-buffered LDS, one barrier
// per 32-deep K step; X rows [0,K0) from X0, [K0,Ktot) from X1 (concat).
// Block: 64x64 tile, 128 threads (4 waves), wave = 16 rows x 64 cols.
// ---------------------------------------------------------------------------
__global__ __launch_bounds__(128) void gemm_k(
    const _Float16* __restrict__ W16, const float* __restrict__ bias,
    const float* __restrict__ bn_gamma, const float* __restrict__ bn_beta,
    const float* __restrict__ bn_mean, const float* __restrict__ bn_var,
    const float* __restrict__ X0, int K0,
    const float* __restrict__ X1,
    const float* __restrict__ resid,
    float* __restrict__ Y, int M, int Ktot, int Ncols)
{
  alignas(16) __shared__ _Float16 As[2][64 * G_LD];
  alignas(16) __shared__ _Float16 Bs[2][64 * G_LD];
  const int b = blockIdx.z;
  const int nblk = blockIdx.x * 64;
  const int mblk = blockIdx.y * 64;
  const int tid = threadIdx.x, lane = tid & 31, wid = tid >> 5;
  const int lr = lane & 15, hi = lane >> 4;
  const int K1 = Ktot - K0;
  const float* x0 = X0 + (size_t)b * K0 * Ncols;
  const float* x1 = X1 ? (X1 + (size_t)b * K1 * Ncols) : nullptr;

  // Weight tile (64 rows x 32 k halves) via async DMA: 256 16B chunks, 2/thread
  auto stageA = [&](int kb, int bsel) {
#pragma unroll
    for (int it = 0; it < 2; ++it) {
      int c = tid + it * 128;
      int m = c >> 2;
      int i8 = (c & 3) * 8;
      async_g2l_b128(&As[bsel][m * G_LD + i8],
                     &W16[(size_t)(mblk + m) * Ktot + kb + i8]);
    }
  };
  // Activation tile transposed [n][k] with fp32->fp16 convert
  auto stageB = [&](int kb, int bsel) {
    for (int idx = tid; idx < 32 * 64; idx += 128) {
      int n = idx & 63, i = idx >> 6;
      int krow = kb + i;
      float val = (krow < K0) ? x0[(size_t)krow * Ncols + nblk + n]
                              : x1[(size_t)(krow - K0) * Ncols + nblk + n];
      Bs[bsel][n * G_LD + i] = (_Float16)val;
    }
  };

  v8f zero = {};
  v8f acc[4] = {zero, zero, zero, zero};

  stageA(0, 0);
  stageB(0, 0);
  wait_async0();
  __syncthreads();

  const int nk = Ktot >> 5;
  for (int t = 0; t < nk; ++t) {
    const int bsel = t & 1;
    if (t + 1 < nk) {            // stage next tile; DMA overlaps this tile's WMMAs
      stageA((t + 1) * 32, bsel ^ 1);
      stageB((t + 1) * 32, bsel ^ 1);
    }
    if (t + 2 < nk) {            // prefetch t+2 activations into L2
      int krow = (t + 2) * 32 + (tid >> 6);
      const float* pf = (krow < K0) ? &x0[(size_t)krow * Ncols + nblk + (tid & 63)]
                                    : &x1[(size_t)(krow - K0) * Ncols + nblk + (tid & 63)];
      __builtin_prefetch(pf, 0, 1);
    }
    v16h a = ld_frag_a(As[bsel], G_LD, wid * 16, 0, lane);
    v16h bf[4];
#pragma unroll
    for (int j = 0; j < 4; ++j) bf[j] = ld_frag_b(Bs[bsel], G_LD, j * 16, 0, lane);
#pragma unroll
    for (int j = 0; j < 4; ++j) acc[j] = WMMA_F16(a, bf[j], acc[j]);
    wait_async0();
    __syncthreads();
  }

  // Epilogue: bias (+ folded BN+ReLU) (+ residual)
  const int orow0 = mblk + wid * 16 + hi * 8;
  float badd[8], scl[8], sft[8];
#pragma unroll
  for (int r = 0; r < 8; ++r) {
    int o = orow0 + r;
    badd[r] = bias[o];
    if (bn_gamma) {
      float inv = rsqrtf(bn_var[o] + 1e-5f);
      scl[r] = bn_gamma[o] * inv;
      sft[r] = bn_beta[o] - bn_mean[o] * scl[r];
    }
  }
  float* yb = Y + (size_t)b * M * Ncols;
  const float* rb = resid ? (resid + (size_t)b * M * Ncols) : nullptr;
#pragma unroll
  for (int j = 0; j < 4; ++j) {
    int n = nblk + j * 16 + lr;
#pragma unroll
    for (int r = 0; r < 8; ++r) {
      int o = orow0 + r;
      float val = acc[j][r] + badd[r];
      if (bn_gamma) val = fmaxf(val * scl[r] + sft[r], 0.f);
      if (rb) val += rb[(size_t)o * Ncols + n];
      yb[(size_t)o * Ncols + n] = val;
    }
  }
}

// ---------------------------------------------------------------------------
// Flash attention per (batch, head, 64-query block). q/k/v: [B,256,N] fp32,
// channel d = e*4 + h (e = head dim 0..63). K/V tiles double-buffered; the
// N x N score matrix only ever exists in WMMA accumulators.
// ---------------------------------------------------------------------------
__global__ __launch_bounds__(128) void attn_k(const float* __restrict__ q,
                                              const float* __restrict__ k,
                                              const float* __restrict__ v,
                                              float* __restrict__ msg, int Nn)
{
  alignas(16) __shared__ _Float16 Qs[64 * AT_LD];     // [query][e]
  alignas(16) __shared__ _Float16 Ks[2][64 * AT_LD];  // [key][e]
  alignas(16) __shared__ _Float16 Vs[2][64 * AT_LD];  // [e][key]
  alignas(16) __shared__ _Float16 Ps[64 * AT_LD];     // [query][key], wave-private bands
  const int b = blockIdx.z, h = blockIdx.y, qb0 = blockIdx.x * 64;
  const size_t bo = (size_t)b * 256 * Nn;
  const float* qp = q + bo;
  const float* kp = k + bo;
  const float* vp = v + bo;
  float* mp = msg + bo;
  const int tid = threadIdx.x, lane = tid & 31, wid = tid >> 5;
  const int lr = lane & 15, hi = lane >> 4;

  auto stageKV = [&](int kb0, int bsel) {
    for (int idx = tid; idx < 64 * 64; idx += 128) {
      int e = idx >> 6, n = idx & 63;
      size_t g = (size_t)(e * 4 + h) * Nn + kb0 + n;
      Ks[bsel][n * AT_LD + e] = (_Float16)kp[g];
      Vs[bsel][e * AT_LD + n] = (_Float16)vp[g];
    }
  };

  for (int idx = tid; idx < 64 * 64; idx += 128) {
    int e = idx >> 6, n = idx & 63;
    Qs[n * AT_LD + e] = (_Float16)qp[(size_t)(e * 4 + h) * Nn + qb0 + n];
  }
  stageKV(0, 0);

  v8f zero = {};
  v8f o[4] = {zero, zero, zero, zero};
  float m_run[8], l_run[8];
#pragma unroll
  for (int r = 0; r < 8; ++r) { m_run[r] = -1e30f; l_run[r] = 0.f; }
  __syncthreads();

  const int nkb = Nn >> 6;
  for (int t = 0; t < nkb; ++t) {
    const int bsel = t & 1;
    if (t + 1 < nkb) stageKV((t + 1) * 64, bsel ^ 1);  // overlaps this block's WMMAs

    // S = (Q K^T) / 8 : M=query, K=e(64), N=key
    v8f s[4] = {zero, zero, zero, zero};
#pragma unroll
    for (int ks = 0; ks < 64; ks += 32) {
      v16h a = ld_frag_a(Qs, AT_LD, wid * 16, ks, lane);
      v16h bf[4];
#pragma unroll
      for (int j = 0; j < 4; ++j) bf[j] = ld_frag_b(Ks[bsel], AT_LD, j * 16, ks, lane);
#pragma unroll
      for (int j = 0; j < 4; ++j) s[j] = WMMA_F16(a, bf[j], s[j]);
    }

    // Online softmax (rows: M = r + 8*hi; cols across 16 lanes x 4 tiles)
    float p[4][8];
#pragma unroll
    for (int r = 0; r < 8; ++r) {
      float rm = -1e30f;
#pragma unroll
      for (int j = 0; j < 4; ++j) rm = fmaxf(rm, s[j][r] * 0.125f);
#pragma unroll
      for (int off = 8; off > 0; off >>= 1) rm = fmaxf(rm, __shfl_xor(rm, off, 16));
      float nm = fmaxf(m_run[r], rm);
      float corr = __expf(m_run[r] - nm);
      m_run[r] = nm;
      float rs = 0.f;
#pragma unroll
      for (int j = 0; j < 4; ++j) {
        float pv = __expf(s[j][r] * 0.125f - nm);
        p[j][r] = pv;
        rs += pv;
      }
#pragma unroll
      for (int off = 8; off > 0; off >>= 1) rs += __shfl_xor(rs, off, 16);
      l_run[r] = l_run[r] * corr + rs;
#pragma unroll
      for (int j = 0; j < 4; ++j) o[j][r] *= corr;
    }

    // P (C-layout regs) -> LDS [q][key] so it reloads as an A fragment
#pragma unroll
    for (int j = 0; j < 4; ++j)
#pragma unroll
      for (int r = 0; r < 8; ++r)
        Ps[(wid * 16 + hi * 8 + r) * AT_LD + j * 16 + lr] = (_Float16)p[j][r];

    // O += P V^T : M=query, K=key(64), N=e
#pragma unroll
    for (int ks = 0; ks < 64; ks += 32) {
      v16h a = ld_frag_a(Ps, AT_LD, wid * 16, ks, lane);
      v16h bf[4];
#pragma unroll
      for (int j = 0; j < 4; ++j) bf[j] = ld_frag_b(Vs[bsel], AT_LD, j * 16, ks, lane);
#pragma unroll
      for (int j = 0; j < 4; ++j) o[j] = WMMA_F16(a, bf[j], o[j]);
    }
    __syncthreads();
  }

#pragma unroll
  for (int j = 0; j < 4; ++j) {
    int e = j * 16 + lr;
#pragma unroll
    for (int r = 0; r < 8; ++r) {
      int qi = qb0 + wid * 16 + hi * 8 + r;
      mp[(size_t)(e * 4 + h) * Nn + qi] = o[j][r] / l_run[r];
    }
  }
}

// ---------------------------------------------------------------------------
// Score GEMM: out[b,n,m] = (mdesc0^T mdesc1)/16 into d_out (ld = 1025)
// ---------------------------------------------------------------------------
__global__ __launch_bounds__(128) void score_k(const float* __restrict__ A,
                                               const float* __restrict__ Bm,
                                               float* __restrict__ out, int Dk, int Nn)
{
  alignas(16) __shared__ _Float16 As[64 * G_LD]; // [n_row][d]
  alignas(16) __shared__ _Float16 Bs[64 * G_LD]; // [m_col][d]
  const int b = blockIdx.z;
  const int rblk = blockIdx.y * 64;
  const int cblk = blockIdx.x * 64;
  const int tid = threadIdx.x, lane = tid & 31, wid = tid >> 5;
  const int lr = lane & 15, hi = lane >> 4;
  const float* ab = A + (size_t)b * Dk * Nn;
  const float* bb = Bm + (size_t)b * Dk * Nn;
  v8f zero = {};
  v8f acc[4] = {zero, zero, zero, zero};
  for (int kb = 0; kb < Dk; kb += 32) {
    for (int idx = tid; idx < 32 * 64; idx += 128) {
      int n = idx & 63, i = idx >> 6;
      As[n * G_LD + i] = (_Float16)ab[(size_t)(kb + i) * Nn + rblk + n];
      Bs[n * G_LD + i] = (_Float16)bb[(size_t)(kb + i) * Nn + cblk + n];
    }
    __syncthreads();
    v16h a = ld_frag_a(As, G_LD, wid * 16, 0, lane);
    v16h bf[4];
#pragma unroll
    for (int j = 0; j < 4; ++j) bf[j] = ld_frag_b(Bs, G_LD, j * 16, 0, lane);
#pragma unroll
    for (int j = 0; j < 4; ++j) acc[j] = WMMA_F16(a, bf[j], acc[j]);
    __syncthreads();
  }
  float* ob = out + (size_t)b * 1025 * 1025;
#pragma unroll
  for (int j = 0; j < 4; ++j) {
    int c = cblk + j * 16 + lr;
#pragma unroll
    for (int r = 0; r < 8; ++r) {
      int rr = rblk + wid * 16 + hi * 8 + r;
      ob[(size_t)rr * 1025 + c] = acc[j][r] * 0.0625f;
    }
  }
}

// ---------------------------------------------------------------------------
// Keypoint encoder: per-point MLP [3,32,64,128,256,256], BN+ReLU except last;
// writes d = desc + enc.
// ---------------------------------------------------------------------------
struct KencParams {
  const float* w[5];
  const float* bsc[5];
  const float* gamma[4];
  const float* beta[4];
  const float* mean[4];
  const float* var[4];
};

__global__ __launch_bounds__(256) void kenc_k(KencParams kp,
    const float* __restrict__ kpts0, const float* __restrict__ kpts1,
    const float* __restrict__ sc0, const float* __restrict__ sc1,
    const float* __restrict__ desc0, const float* __restrict__ desc1,
    float* __restrict__ d0, float* __restrict__ d1, int Nn)
{
  __shared__ float buf[2][256];
  const int set = blockIdx.y;
  const int b = blockIdx.x / Nn, n = blockIdx.x % Nn;
  const float* kpts = set ? kpts1 : kpts0;
  const float* sc = set ? sc1 : sc0;
  const float* desc = set ? desc1 : desc0;
  float* dout = set ? d1 : d0;
  const int t = threadIdx.x;
  if (t == 0) buf[0][0] = (kpts[((size_t)b * Nn + n) * 2 + 0] - 320.f) / 320.f;
  if (t == 1) buf[0][1] = (kpts[((size_t)b * Nn + n) * 2 + 1] - 240.f) / 240.f;
  if (t == 2) buf[0][2] = sc[(size_t)b * Nn + n];
  __syncthreads();
  const int cin[5] = {3, 32, 64, 128, 256};
  const int cout[5] = {32, 64, 128, 256, 256};
  int cur = 0;
  for (int l = 0; l < 5; ++l) {
    if (t < cout[l]) {
      float acc = kp.bsc[l][t];
      const float* wr = kp.w[l] + (size_t)t * cin[l];
      for (int i = 0; i < cin[l]; ++i) acc += wr[i] * buf[cur][i];
      if (l < 4) {
        float inv = rsqrtf(kp.var[l][t] + 1e-5f);
        acc = kp.gamma[l][t] * (acc - kp.mean[l][t]) * inv + kp.beta[l][t];
        acc = fmaxf(acc, 0.f);
      }
      buf[1 - cur][t] = acc;
    }
    __syncthreads();
    cur ^= 1;
  }
  size_t gi = (size_t)b * 256 * Nn + (size_t)t * Nn + n;
  dout[gi] = buf[cur][t] + desc[gi];
}

// ---------------------------------------------------------------------------
// Sinkhorn in log space, Z (couplings) lives in d_out. NORM = -log(2048).
// ---------------------------------------------------------------------------
#define SK_NORM (-7.6246189861593985f)
#define SK_NLOG2 (-0.6931471805599453f)
#define SK_LOG2048 (7.6246189861593985f)

__global__ void zero2_k(float* __restrict__ a, float* __restrict__ b, int n)
{
  int i = blockIdx.x * blockDim.x + threadIdx.x;
  if (i < n) { a[i] = 0.f; b[i] = 0.f; }
}

__global__ void bins_k(const float* __restrict__ alpha, float* __restrict__ out)
{
  int idx = blockIdx.x * blockDim.x + threadIdx.x;
  if (idx >= 4 * 1025) return;
  int b = idx / 1025, i = idx % 1025;
  float a = *alpha;
  size_t base = (size_t)b * 1025 * 1025;
  out[base + (size_t)i * 1025 + 1024] = a;
  out[base + (size_t)1024 * 1025 + i] = a;
}

__global__ __launch_bounds__(256) void sink_row_k(const float* __restrict__ Z,
    const float* __restrict__ vv, float* __restrict__ uu, int Nn)
{
  const int b = blockIdx.y, n = blockIdx.x, tid = threadIdx.x;
  const float* zr = Z + ((size_t)b * 1025 + n) * 1025;
  const float* vb = vv + b * 1025;
  __shared__ float sred[256];
  float lmax = -1e30f;
  for (int m = tid; m < 1025; m += 256) lmax = fmaxf(lmax, zr[m] + vb[m]);
  sred[tid] = lmax; __syncthreads();
  for (int s = 128; s > 0; s >>= 1) { if (tid < s) sred[tid] = fmaxf(sred[tid], sred[tid + s]); __syncthreads(); }
  float gmax = sred[0]; __syncthreads();
  float lsum = 0.f;
  for (int m = tid; m < 1025; m += 256) lsum += __expf(zr[m] + vb[m] - gmax);
  sred[tid] = lsum; __syncthreads();
  for (int s = 128; s > 0; s >>= 1) { if (tid < s) sred[tid] += sred[tid + s]; __syncthreads(); }
  if (tid == 0) {
    float logmu = (n < Nn) ? SK_NORM : SK_NLOG2;
    uu[b * 1025 + n] = logmu - (gmax + __logf(sred[0]));
  }
}

__global__ __launch_bounds__(256) void sink_col_k(const float* __restrict__ Z,
    const float* __restrict__ uu, float* __restrict__ vv, int Nn)
{
  const int b = blockIdx.y, m = blockIdx.x, tid = threadIdx.x;
  const float* zc = Z + (size_t)b * 1025 * 1025 + m;
  const float* ub = uu + b * 1025;
  __shared__ float sred[256];
  float lmax = -1e30f;
  for (int n = tid; n < 1025; n += 256) lmax = fmaxf(lmax, zc[(size_t)n * 1025] + ub[n]);
  sred[tid] = lmax; __syncthreads();
  for (int s = 128; s > 0; s >>= 1) { if (tid < s) sred[tid] = fmaxf(sred[tid], sred[tid + s]); __syncthreads(); }
  float gmax = sred[0]; __syncthreads();
  float lsum = 0.f;
  for (int n = tid; n < 1025; n += 256) lsum += __expf(zc[(size_t)n * 1025] + ub[n] - gmax);
  sred[tid] = lsum; __syncthreads();
  for (int s = 128; s > 0; s >>= 1) { if (tid < s) sred[tid] += sred[tid + s]; __syncthreads(); }
  if (tid == 0) {
    float lognu = (m < Nn) ? SK_NORM : SK_NLOG2;
    vv[b * 1025 + m] = lognu - (gmax + __logf(sred[0]));
  }
}

__global__ void final_k(float* __restrict__ out, const float* __restrict__ uu,
                        const float* __restrict__ vv, int total)
{
  int i = blockIdx.x * blockDim.x + threadIdx.x;
  if (i >= total) return;
  int b = i / (1025 * 1025);
  int rem = i - b * 1025 * 1025;
  int n = rem / 1025, m = rem - n * 1025;
  out[i] = out[i] + uu[b * 1025 + n] + vv[b * 1025 + m] + SK_LOG2048;
}

// ---------------------------------------------------------------------------
// Host-side orchestration
// ---------------------------------------------------------------------------
extern "C" void kernel_launch(void* const* d_in, const int* in_sizes, int n_in,
                              void* d_out, int out_size, void* d_ws, size_t ws_size,
                              hipStream_t stream)
{
  (void)in_sizes; (void)n_in; (void)ws_size;
  const int Bn = 4, Dn = 256, Nn = 1024;
  auto F = [&](int i) { return (const float*)d_in[i]; };

  const float* desc0 = F(0);
  const float* desc1 = F(1);
  const float* kpts0 = F(2);
  const float* kpts1 = F(3);
  const float* sc0 = F(4);
  const float* sc1 = F(5);
  // Param pytree (sorted dict keys): bin, final{b,w}, gnn[18]{k,merge,mlp,q,v}, kenc[5]
  const float* alpha = F(6);
  const int FINAL_B = 7, FINAL_W = 8, GNN0 = 9, KENC0 = 9 + 18 * 16;

  // Workspace carve-up: fp32 activations, then fp16 weight area
  size_t off = 0;
  auto alloc = [&](size_t nf) { float* p = (float*)d_ws + off; off += nf; return p; };
  const size_t BDN = (size_t)Bn * Dn * Nn;
  float* d0 = alloc(BDN);
  float* d1 = alloc(BDN);
  float* nd0 = alloc(BDN);
  float* nd1 = alloc(BDN);
  float* qbuf = alloc(BDN);
  float* kbuf = alloc(BDN);
  float* vbuf = alloc(BDN);
  float* msgb = alloc(BDN);
  float* mrg = alloc(BDN);
  float* hid = alloc((size_t)Bn * 512 * Nn);
  float* md0 = alloc(BDN);
  float* md1 = alloc(BDN);
  float* uvec = alloc((size_t)Bn * 1025);
  float* vvec = alloc((size_t)Bn * 1025);

  _Float16* hb = (_Float16*)((float*)d_ws + off);
  size_t hoff = 0;
  auto alloch = [&](size_t nh) { _Float16* p = hb + hoff; hoff += nh; return p; };
  auto cvt = [&](const float* src, _Float16* dst, int n) {
    cvt16_k<<<dim3((n + 255) / 256), dim3(256), 0, stream>>>(src, dst, n);
  };

  struct LayerW { _Float16 *q, *k, *v, *m, *c1, *c2; };
  LayerW lw[18];
  for (int L = 0; L < 18; ++L) {
    int gb = GNN0 + L * 16;
    lw[L].k = alloch(65536);   cvt(F(gb + 1), lw[L].k, 65536);
    lw[L].m = alloch(65536);   cvt(F(gb + 3), lw[L].m, 65536);
    lw[L].c1 = alloch(262144); cvt(F(gb + 9), lw[L].c1, 262144);
    lw[L].c2 = alloch(131072); cvt(F(gb + 11), lw[L].c2, 131072);
    lw[L].q = alloch(65536);   cvt(F(gb + 13), lw[L].q, 65536);
    lw[L].v = alloch(65536);   cvt(F(gb + 15), lw[L].v, 65536);
  }
  _Float16* finW = alloch(65536);
  cvt(F(FINAL_W), finW, 65536);

  // Keypoint encoder
  KencParams kp;
  for (int l = 0; l < 4; ++l) {
    int base = KENC0 + l * 6;
    kp.beta[l] = F(base + 0);
    kp.gamma[l] = F(base + 1);
    kp.mean[l] = F(base + 2);
    kp.var[l] = F(base + 3);
    kp.bsc[l] = F(base + 4);
    kp.w[l] = F(base + 5);
  }
  kp.bsc[4] = F(KENC0 + 24);
  kp.w[4] = F(KENC0 + 25);
  kenc_k<<<dim3(Bn * Nn, 2, 1), dim3(256), 0, stream>>>(
      kp, kpts0, kpts1, sc0, sc1, desc0, desc1, d0, d1, Nn);

  auto gemm = [&](const _Float16* W16, const float* bias,
                  const float* g, const float* be, const float* me, const float* va,
                  const float* X0, int K0, const float* X1, int Ktot,
                  const float* resid, float* Y, int M) {
    gemm_k<<<dim3(Nn / 64, M / 64, Bn), dim3(128), 0, stream>>>(
        W16, bias, g, be, me, va, X0, K0, X1, resid, Y, M, Ktot, Nn);
  };

  // 18 GNN layers: 'self','cross' alternating
  for (int L = 0; L < 18; ++L) {
    int gb = GNN0 + L * 16;
    const float* kB = F(gb + 0);
    const float* mB = F(gb + 2);
    const float* bnBe = F(gb + 4); const float* bnGa = F(gb + 5);
    const float* bnMe = F(gb + 6); const float* bnVa = F(gb + 7);
    const float* c1B = F(gb + 8);
    const float* c2B = F(gb + 10);
    const float* qB = F(gb + 12);
    const float* vB = F(gb + 14);
    const bool cross = (L & 1);

    for (int side = 0; side < 2; ++side) {
      const float* x = side ? d1 : d0;
      const float* other = side ? d0 : d1;
      const float* src = cross ? other : x;
      float* outd = side ? nd1 : nd0;

      gemm(lw[L].q, qB, 0, 0, 0, 0, x, Dn, nullptr, Dn, nullptr, qbuf, Dn);
      gemm(lw[L].k, kB, 0, 0, 0, 0, src, Dn, nullptr, Dn, nullptr, kbuf, Dn);
      gemm(lw[L].v, vB, 0, 0, 0, 0, src, Dn, nullptr, Dn, nullptr, vbuf, Dn);
      attn_k<<<dim3(Nn / 64, 4, Bn), dim3(128), 0, stream>>>(qbuf, kbuf, vbuf, msgb, Nn);
      gemm(lw[L].m, mB, 0, 0, 0, 0, msgb, Dn, nullptr, Dn, nullptr, mrg, Dn);
      // MLP layer 1: concat([x, merged]) -> 512, BN+ReLU
      gemm(lw[L].c1, c1B, bnGa, bnBe, bnMe, bnVa, x, Dn, mrg, 512, nullptr, hid, 512);
      // MLP layer 2: 512 -> 256, residual add with x
      gemm(lw[L].c2, c2B, 0, 0, 0, 0, hid, 512, nullptr, 512, x, outd, Dn);
    }
    float* t0 = d0; d0 = nd0; nd0 = t0;
    float* t1 = d1; d1 = nd1; nd1 = t1;
  }

  // Final projection + score matrix (into d_out, ld=1025) + bins
  gemm(finW, F(FINAL_B), 0, 0, 0, 0, d0, Dn, nullptr, Dn, nullptr, md0, Dn);
  gemm(finW, F(FINAL_B), 0, 0, 0, 0, d1, Dn, nullptr, Dn, nullptr, md1, Dn);
  score_k<<<dim3(Nn / 64, Nn / 64, Bn), dim3(128), 0, stream>>>(md0, md1, (float*)d_out, Dn, Nn);
  bins_k<<<dim3((Bn * 1025 + 255) / 256), dim3(256), 0, stream>>>(alpha, (float*)d_out);

  // Sinkhorn (20 iters) + finalize
  zero2_k<<<dim3((Bn * 1025 + 255) / 256), dim3(256), 0, stream>>>(uvec, vvec, Bn * 1025);
  for (int it = 0; it < 20; ++it) {
    sink_row_k<<<dim3(1025, Bn), dim3(256), 0, stream>>>((const float*)d_out, vvec, uvec, Nn);
    sink_col_k<<<dim3(1025, Bn), dim3(256), 0, stream>>>((const float*)d_out, uvec, vvec, Nn);
  }
  final_k<<<dim3((out_size + 255) / 256), dim3(256), 0, stream>>>(
      (float*)d_out, uvec, vvec, out_size);
}
